// Attention2_29291676959366
// MI455X (gfx1250) — compile-verified, matrix operand order checked by reading, not run on hardware
//
#include <hip/hip_runtime.h>
#include <hip/hip_bf16.h>

typedef __bf16 bf16_t;
typedef __attribute__((ext_vector_type(16))) __bf16 v16bf;
typedef __attribute__((ext_vector_type(8)))  __bf16 v8bf;
typedef __attribute__((ext_vector_type(8)))  float   v8f;

// ---------------------------------------------------------------------------
// Fragment loader for 16-bit A/B operands of v_wmma_*_16x16x32_bf16.
// Source is row-major [rows][ld] bf16. Per CDNA5 ISA (7.12.2, 16-bit A 16x32):
//   lanes 0-15 : VGPR0-3 hold K=0..7,  VGPR4-7 hold K=16..23
//   lanes 16-31: VGPR0-3 hold K=8..15, VGPR4-7 hold K=24..31
// B (32x16) is the mirrored layout with lane = N. Both reduce to two
// contiguous 16-byte loads per lane -> global/ds _load_b128.
// ---------------------------------------------------------------------------
static __device__ __forceinline__ v16bf load_frag(const bf16_t* base, int ld,
                                                  int row0, int k0) {
  const int lane = (int)(threadIdx.x & 31u);
  const int half = lane >> 4;
  const bf16_t* p = base + (long)(row0 + (lane & 15)) * ld + k0 + half * 8;
  v8bf lo = *(const v8bf*)(p);
  v8bf hi = *(const v8bf*)(p + 16);
  v16bf out;
#pragma unroll
  for (int i = 0; i < 8; ++i) { out[i] = lo[i]; out[i + 8] = hi[i]; }
  return out;
}

#define WMMA_BF16(A, B, C) \
  __builtin_amdgcn_wmma_f32_16x16x32_bf16(false, (A), false, (B), (short)0, (C), false, false)

// ---------------------------------------------------------------------------
// Generic bf16 GEMM:  out[b] = A[b] (MxK, bf16) * BT^T (BT is [N][K] bf16)
// Epilogue: +bias (optional); either bf16 store, or fp32 "gamma*acc + resid".
// Block: 256 threads (8 waves). Tile 128(M) x 64(N); wave tile 32x32.
// ---------------------------------------------------------------------------
__global__ __launch_bounds__(256) void gemm_bf16_kernel(
    const bf16_t* __restrict__ A, long bsA,
    const bf16_t* __restrict__ BT,
    const float* __restrict__ bias,
    bf16_t* __restrict__ outB, float* __restrict__ outF, long bsO,
    const float* __restrict__ resid, const float* __restrict__ gamma,
    int N, int K) {
  const int wid  = (int)(threadIdx.x >> 5);
  const int lane = (int)(threadIdx.x & 31u);
  const int half = lane >> 4;
  const int ln   = lane & 15;
  const int mRow = blockIdx.x * 128 + (wid >> 1) * 32;
  const int nCol = blockIdx.y * 64 + (wid & 1) * 32;
  const bf16_t* Ab = A + (long)blockIdx.z * bsA;

  v8f acc[2][2] = {};
  for (int k = 0; k < K; k += 32) {
    v16bf a0 = load_frag(Ab, K, mRow, k);
    v16bf a1 = load_frag(Ab, K, mRow + 16, k);
    v16bf b0 = load_frag(BT, K, nCol, k);
    v16bf b1 = load_frag(BT, K, nCol + 16, k);
    acc[0][0] = WMMA_BF16(a0, b0, acc[0][0]);
    acc[0][1] = WMMA_BF16(a0, b1, acc[0][1]);
    acc[1][0] = WMMA_BF16(a1, b0, acc[1][0]);
    acc[1][1] = WMMA_BF16(a1, b1, acc[1][1]);
  }

  const float gm = gamma ? gamma[0] : 0.0f;
#pragma unroll
  for (int i = 0; i < 2; ++i) {
#pragma unroll
    for (int j = 0; j < 2; ++j) {
      const int col = nCol + j * 16 + ln;
      const float bv = bias ? bias[col] : 0.0f;
#pragma unroll
      for (int r = 0; r < 8; ++r) {
        const int row = mRow + i * 16 + r + half * 8;  // C-layout row mapping
        const float v = acc[i][j][r] + bv;
        const long off = (long)blockIdx.z * bsO + (long)row * N + col;
        if (outF) outF[off] = gm * v + resid[off];
        else      outB[off] = (bf16_t)v;
      }
    }
  }
}

// ---------------------------------------------------------------------------
// Flash attention: o = softmax(g f^T) h   (no scaling, per reference)
// G:[B][Nq][64]  F:[B][Nk][64]  HT:[B][Dv][Nk]  O:[B][Nq][Dv]
// Block: 256 thr = 8 waves = 4 query-groups(16 rows) x 2 vcol-groups(128 cols)
// ---------------------------------------------------------------------------
__global__ __launch_bounds__(256) void attn_kernel(
    const bf16_t* __restrict__ G, const bf16_t* __restrict__ F,
    const bf16_t* __restrict__ HT, bf16_t* __restrict__ O,
    int Nq, int Nk, int Dv) {
  __shared__ bf16_t pbuf[4][16][64];  // per query-group probability tile

  const int wid  = (int)(threadIdx.x >> 5);
  const int lane = (int)(threadIdx.x & 31u);
  const int half = lane >> 4;
  const int ln   = lane & 15;
  const int qg = wid >> 1;
  const int vg = wid & 1;
  const int b  = blockIdx.y;
  const int q0 = blockIdx.x * 64 + qg * 16;

  const bf16_t* Gb = G + (long)b * Nq * 64;
  const bf16_t* Fb = F + (long)b * Nk * 64;
  const bf16_t* Hb = HT + (long)b * Dv * Nk;
  bf16_t* Ob = O + (long)b * Nq * Dv;

  const v16bf ga0 = load_frag(Gb, 64, q0, 0);
  const v16bf ga1 = load_frag(Gb, 64, q0, 32);

  v8f oacc[8] = {};
  float mrow[8], lrow[8];
#pragma unroll
  for (int r = 0; r < 8; ++r) { mrow[r] = -1e30f; lrow[r] = 0.0f; }

  for (int kt = 0; kt < Nk; kt += 64) {
    // ---- scores: s[16 x 64] = g_tile . f_tile^T (f rows are contiguous K)
    v8f s[4] = {};
#pragma unroll
    for (int nt = 0; nt < 4; ++nt) {
      v16bf fb0 = load_frag(Fb, 64, kt + nt * 16, 0);
      v16bf fb1 = load_frag(Fb, 64, kt + nt * 16, 32);
      s[nt] = WMMA_BF16(ga0, fb0, s[nt]);
      s[nt] = WMMA_BF16(ga1, fb1, s[nt]);
    }

    // ---- streaming softmax (rows live across the 16 lanes of each half)
    float sc[8];
#pragma unroll
    for (int r = 0; r < 8; ++r) {
      float t = fmaxf(fmaxf(s[0][r], s[1][r]), fmaxf(s[2][r], s[3][r]));
      t = fmaxf(t, __shfl_xor(t, 1, 32));
      t = fmaxf(t, __shfl_xor(t, 2, 32));
      t = fmaxf(t, __shfl_xor(t, 4, 32));
      t = fmaxf(t, __shfl_xor(t, 8, 32));
      const float mn = fmaxf(mrow[r], t);
      sc[r] = __expf(mrow[r] - mn);
      float rs = 0.0f;
#pragma unroll
      for (int nt = 0; nt < 4; ++nt) {
        const float e = __expf(s[nt][r] - mn);
        s[nt][r] = e;
        rs += e;
      }
      rs += __shfl_xor(rs, 1, 32);
      rs += __shfl_xor(rs, 2, 32);
      rs += __shfl_xor(rs, 4, 32);
      rs += __shfl_xor(rs, 8, 32);
      lrow[r] = lrow[r] * sc[r] + rs;
      mrow[r] = mn;
    }

    // ---- transpose p from C-layout to A-layout via LDS
    __syncthreads();  // previous-iteration readers of pbuf are done
    if (vg == 0) {
#pragma unroll
      for (int nt = 0; nt < 4; ++nt)
#pragma unroll
        for (int r = 0; r < 8; ++r)
          pbuf[qg][r + half * 8][nt * 16 + ln] = (bf16_t)s[nt][r];
    }
    __syncthreads();
    const v16bf pa0 = load_frag(&pbuf[qg][0][0], 64, 0, 0);
    const v16bf pa1 = load_frag(&pbuf[qg][0][0], 64, 0, 32);

    // ---- rescale running O, accumulate p . h  (HT rows contiguous in keys)
#pragma unroll
    for (int vt = 0; vt < 8; ++vt)
#pragma unroll
      for (int r = 0; r < 8; ++r) oacc[vt][r] *= sc[r];
#pragma unroll
    for (int vt = 0; vt < 8; ++vt) {
      v16bf hb0 = load_frag(Hb, Nk, vg * 128 + vt * 16, kt);
      v16bf hb1 = load_frag(Hb, Nk, vg * 128 + vt * 16, kt + 32);
      oacc[vt] = WMMA_BF16(pa0, hb0, oacc[vt]);
      oacc[vt] = WMMA_BF16(pa1, hb1, oacc[vt]);
    }
  }

  // ---- normalize and store
#pragma unroll
  for (int vt = 0; vt < 8; ++vt) {
    const int col = vg * 128 + vt * 16 + ln;
#pragma unroll
    for (int r = 0; r < 8; ++r) {
      const int row = q0 + r + half * 8;
      Ob[(long)row * Dv + col] = (bf16_t)(oacc[vt][r] / lrow[r]);
    }
  }
}

// ---------------------------------------------------------------------------
// Elementwise helpers
// ---------------------------------------------------------------------------
__global__ void cvt_bf16_kernel(const float* __restrict__ in,
                                bf16_t* __restrict__ out, long n) {
  long i = blockIdx.x * (long)blockDim.x + threadIdx.x;
  const long stride = (long)gridDim.x * blockDim.x;
  for (; i < n; i += stride) out[i] = (bf16_t)in[i];
}

// [R][C] f32 -> [C][R] bf16 (weight convert+transpose)
__global__ void cvt_T_bf16_kernel(const float* __restrict__ in,
                                  bf16_t* __restrict__ out, int R, int C) {
  long i = blockIdx.x * (long)blockDim.x + threadIdx.x;
  const long n = (long)R * C;
  const long stride = (long)gridDim.x * blockDim.x;
  for (; i < n; i += stride) {
    const int r = (int)(i / C);
    const int c = (int)(i % C);
    out[(long)c * R + r] = (bf16_t)in[i];
  }
}

// per-batch [R][C] -> [C][R], bf16 (grid.z = batch)
__global__ void transpose_bf16_kernel(const bf16_t* __restrict__ in,
                                      bf16_t* __restrict__ out, int R, int C) {
  const long n = (long)R * C;
  const bf16_t* ib = in + (long)blockIdx.z * n;
  bf16_t* ob = out + (long)blockIdx.z * n;
  long i = blockIdx.x * (long)blockDim.x + threadIdx.x;
  const long stride = (long)gridDim.x * blockDim.x;
  for (; i < n; i += stride) {
    const int r = (int)(i / C);
    const int c = (int)(i % C);
    ob[(long)c * R + r] = ib[i];
  }
}

// per-batch 2x2 maxpool NHWC, bf16 (grid.z = batch)
__global__ void pool2x2_kernel(const bf16_t* __restrict__ in,
                               bf16_t* __restrict__ out, int H, int W, int C) {
  const long npb = (long)(H / 2) * (W / 2) * C;
  const bf16_t* ib = in + (long)blockIdx.z * H * W * C;
  bf16_t* ob = out + (long)blockIdx.z * npb;
  long i = blockIdx.x * (long)blockDim.x + threadIdx.x;
  const long stride = (long)gridDim.x * blockDim.x;
  for (; i < npb; i += stride) {
    const int c = (int)(i % C);
    const long t = i / C;
    const int w = (int)(t % (W / 2));
    const int h = (int)(t / (W / 2));
    const long base = ((long)(2 * h) * W + 2 * w) * C + c;
    const float a = (float)ib[base];
    const float bb = (float)ib[base + C];
    const float cc = (float)ib[base + (long)W * C];
    const float dd = (float)ib[base + (long)W * C + C];
    ob[i] = (bf16_t)fmaxf(fmaxf(a, bb), fmaxf(cc, dd));
  }
}

// ---------------------------------------------------------------------------
extern "C" void kernel_launch(void* const* d_in, const int* in_sizes, int n_in,
                              void* d_out, int out_size, void* d_ws,
                              size_t ws_size, hipStream_t stream) {
  const float* x      = (const float*)d_in[0];
  const float* kf     = (const float*)d_in[1];
  const float* kg     = (const float*)d_in[2];
  const float* kh     = (const float*)d_in[3];
  const float* ko     = (const float*)d_in[4];
  const float* bias_f = (const float*)d_in[5];
  const float* bias_g = (const float*)d_in[6];
  const float* bias_h = (const float*)d_in[7];
  const float* gamma  = (const float*)d_in[8];
  float* out = (float*)d_out;

  const int B = 8, Hh = 64, Ww = 64, C = 512;
  const int Nq = Hh * Ww;        // 4096
  const int Np = Nq / 4;         // 1024 pooled positions
  const int CF = C / 8;          // 64  (f/g channels)
  const int CH2 = C / 2;         // 256 (h channels)

  char* ws = (char*)d_ws;
  size_t off = 0;
  auto alloc = [&](size_t bytes) -> char* {
    char* p = ws + off;
    off += (bytes + 255) & ~(size_t)255;
    return p;
  };
  bf16_t* xb    = (bf16_t*)alloc((size_t)B * Nq * C * 2);     // x in bf16
  bf16_t* wfT   = (bf16_t*)alloc((size_t)CF * C * 2);         // [64][512]
  bf16_t* wgT   = (bf16_t*)alloc((size_t)CF * C * 2);
  bf16_t* whT   = (bf16_t*)alloc((size_t)CH2 * C * 2);        // [256][512]
  bf16_t* woT   = (bf16_t*)alloc((size_t)C * CH2 * 2);        // [512][256]
  bf16_t* gbuf  = (bf16_t*)alloc((size_t)B * Nq * CF * 2);    // g
  bf16_t* funp  = (bf16_t*)alloc((size_t)B * Nq * CF * 2);    // f pre-pool
  bf16_t* fpool = (bf16_t*)alloc((size_t)B * Np * CF * 2);    // f pooled
  bf16_t* hpool = (bf16_t*)alloc((size_t)B * Np * CH2 * 2);   // h pooled
  bf16_t* hT    = (bf16_t*)alloc((size_t)B * CH2 * Np * 2);   // h^T [Dv][Nk]
  bf16_t* hunp  = (bf16_t*)alloc((size_t)B * Nq * CH2 * 2);   // h pre-pool
  bf16_t* obuf  = hunp;  // reuse: h pre-pool dead after pooling

  // 1) precision conversion (bf16 everywhere the WMMA units eat)
  cvt_bf16_kernel<<<4096, 256, 0, stream>>>(x, xb, (long)B * Nq * C);
  cvt_T_bf16_kernel<<<64, 256, 0, stream>>>(kf, wfT, C, CF);
  cvt_T_bf16_kernel<<<64, 256, 0, stream>>>(kg, wgT, C, CF);
  cvt_T_bf16_kernel<<<256, 256, 0, stream>>>(kh, whT, C, CH2);
  cvt_T_bf16_kernel<<<256, 256, 0, stream>>>(ko, woT, CH2, C);

  // 2) projection GEMMs (1x1 convs)
  gemm_bf16_kernel<<<dim3(Nq / 128, CF / 64, B), 256, 0, stream>>>(
      xb, (long)Nq * C, wgT, bias_g, gbuf, nullptr, (long)Nq * CF,
      nullptr, nullptr, CF, C);
  gemm_bf16_kernel<<<dim3(Nq / 128, CF / 64, B), 256, 0, stream>>>(
      xb, (long)Nq * C, wfT, bias_f, funp, nullptr, (long)Nq * CF,
      nullptr, nullptr, CF, C);
  gemm_bf16_kernel<<<dim3(Nq / 128, CH2 / 64, B), 256, 0, stream>>>(
      xb, (long)Nq * C, whT, bias_h, hunp, nullptr, (long)Nq * CH2,
      nullptr, nullptr, CH2, C);

  // 3) pooling + h transpose (for contiguous-K B fragments in attention)
  pool2x2_kernel<<<dim3(256, 1, B), 256, 0, stream>>>(funp, fpool, Hh, Ww, CF);
  pool2x2_kernel<<<dim3(1024, 1, B), 256, 0, stream>>>(hunp, hpool, Hh, Ww, CH2);
  transpose_bf16_kernel<<<dim3(1024, 1, B), 256, 0, stream>>>(hpool, hT, Np, CH2);

  // 4) fused flash attention
  attn_kernel<<<dim3(Nq / 64, B), 256, 0, stream>>>(gbuf, fpool, hT, obuf,
                                                    Nq, Np, CH2);

  // 5) output projection + gamma*o + x residual (fp32 out)
  gemm_bf16_kernel<<<dim3(Nq / 128, C / 64, B), 256, 0, stream>>>(
      obuf, (long)Nq * CH2, woT, nullptr, nullptr, out, (long)Nq * C,
      x, gamma, C, CH2);
}
